// BarycentricMapping_3281355014643
// MI455X (gfx1250) — matrix operand order repaired and checked by reading, not compiled, not run on hardware
//
#include <hip/hip_runtime.h>
#include <math.h>

#define NN 8192
#define MM 8192
#define DD 128
#define INV_REG 20.0f          // 1/0.05
#define N_ITER 50
#define LOGNM 9.01091331f      // log(8192)
#define CHUNKS 64
#define ROWS_PER_CHUNK (NN / CHUNKS)   // 128

typedef __attribute__((ext_vector_type(16))) __bf16        v16bf;
typedef __attribute__((ext_vector_type(8)))  float         v8f;
typedef __attribute__((ext_vector_type(4)))  unsigned int  v4u;
typedef __attribute__((ext_vector_type(4)))  float         v4f;
typedef __attribute__((ext_vector_type(2)))  float         v2f;

union ABu { v16bf bf; v4u q[2]; };   // 32 bytes, POD

__device__ __forceinline__ unsigned short f2bfu(float f) {
  unsigned u = __builtin_bit_cast(unsigned, f);
  unsigned r = u + 0x7FFFu + ((u >> 16) & 1u);   // round-to-nearest-even
  return (unsigned short)(r >> 16);
}

// Branch-free online logsumexp accumulate: one exp per element.
__device__ __forceinline__ void lse_push(float& m, float& s, float x) {
  float e  = __expf(-fabsf(x - m));      // exp(min-max): always <= 1
  float sg = __fmaf_rn(s, e, 1.0f);      // case x > m
  float sl = s + e;                      // case x <= m
  bool  gt = x > m;
  s = gt ? sg : sl;
  m = gt ? x : m;
}

// Merge two (max, sumexp) partials.
__device__ __forceinline__ void lse_merge(float& m, float& s, float m2, float s2) {
  float mn = fmaxf(m, m2);
  s = s * __expf(m - mn) + s2 * __expf(m2 - mn);
  m = mn;
}

// ---------------------------------------------------------------------------
// K1: bf16 conversion of XP/XQ, transposed XQ, row squared norms, u=v=0
// ---------------------------------------------------------------------------
__global__ __launch_bounds__(128)
void ot_prep(const float* __restrict__ XP, const float* __restrict__ XQ,
             float* __restrict__ xx, float* __restrict__ yy,
             float* __restrict__ u, float* __restrict__ v,
             unsigned short* __restrict__ XPb, unsigned short* __restrict__ XQb,
             unsigned short* __restrict__ XQt) {
  const int b = blockIdx.x;
  const int t = threadIdx.x;
  __shared__ float red[128];
  if (b < NN) {
    float val = XP[(size_t)b * DD + t];
    XPb[(size_t)b * DD + t] = f2bfu(val);
    red[t] = val * val;
    __syncthreads();
    for (int off = 64; off; off >>= 1) { if (t < off) red[t] += red[t + off]; __syncthreads(); }
    if (t == 0) { xx[b] = red[0]; u[b] = 0.0f; }
  } else {
    const int j = b - NN;
    float val = XQ[(size_t)j * DD + t];
    unsigned short h = f2bfu(val);
    XQb[(size_t)j * DD + t] = h;
    XQt[(size_t)t * MM + j] = h;     // transposed copy for mapped-GEMM B frags
    red[t] = val * val;
    __syncthreads();
    for (int off = 64; off; off >>= 1) { if (t < off) red[t] += red[t + off]; __syncthreads(); }
    if (t == 0) { yy[j] = red[0]; v[j] = 0.0f; }
  }
}

// ---------------------------------------------------------------------------
// K2: Mr[i,j] = -max(0, xx[i]+yy[j]-2*XP.XQ^T)/reg  via bf16 WMMA.
// One wave computes a 16x64 strip: 1 A fragment reused by 4 B fragments
// per k-step, 4 k-steps -> 16 WMMAs per wave.
// ---------------------------------------------------------------------------
__global__ __launch_bounds__(256)
void ot_cost_gemm(const unsigned short* __restrict__ XPb,
                  const unsigned short* __restrict__ XQb,
                  const float* __restrict__ xx, const float* __restrict__ yy,
                  float* __restrict__ Mr) {
  const int w     = blockIdx.x * 8 + (threadIdx.x >> 5);
  const int lane  = threadIdx.x & 31;
  const int tcolg = w & (MM / 64 - 1);    // 128 column groups of 64
  const int trow  = w >> 7;               // 512 row tiles of 16
  const int half  = lane >> 4;            // 0 | 1
  const int l15   = lane & 15;
  const int arow  = trow * 16 + l15;
  const int koff  = half * 8;             // unused (A uses krng)
  const int krng  = half * 16;            // B: K range

  v8f acc[4] = {};
#pragma unroll
  for (int k0 = 0; k0 < DD; k0 += 32) {
    ABu a;
    const unsigned short* ap = XPb + (size_t)arow * DD + k0;
    a.q[0] = *(const v4u*)(ap + krng);
    a.q[1] = *(const v4u*)(ap + krng + 8);
#pragma unroll
    for (int c = 0; c < 4; ++c) {
      ABu b;
      const int bcol = tcolg * 64 + c * 16 + l15;
      const unsigned short* bp = XQb + (size_t)bcol * DD + k0 + krng;
      b.q[0] = *(const v4u*)(bp);
      b.q[1] = *(const v4u*)(bp + 8);
      acc[c] = __builtin_amdgcn_wmma_f32_16x16x32_bf16(false, a.bf, false, b.bf,
                                                       (short)0, acc[c], false, false);
    }
  }
  const int rbase = trow * 16 + half * 8;
  float xr[8];
#pragma unroll
  for (int r = 0; r < 8; ++r) xr[r] = xx[rbase + r];
#pragma unroll
  for (int c = 0; c < 4; ++c) {
    const int col = tcolg * 64 + c * 16 + l15;
    const float yyc = yy[col];
#pragma unroll
    for (int r = 0; r < 8; ++r) {
      float Cij = fmaxf(xr[r] + yyc - 2.0f * acc[c][r], 0.0f);
      Mr[(size_t)(rbase + r) * MM + col] = -INV_REG * Cij;
    }
  }
}

// ---------------------------------------------------------------------------
// K3: column-logsumexp partials. Thread owns 4 consecutive columns (b128
// loads), sweeps a 128-row chunk. Branch-free online lse.
// ---------------------------------------------------------------------------
__global__ __launch_bounds__(256)
void ot_col_partial(const float* __restrict__ Mr, const float* __restrict__ u,
                    float* __restrict__ pm, float* __restrict__ ps) {
  const int j     = (blockIdx.x * 256 + threadIdx.x) * 4;
  const int chunk = blockIdx.y;
  const int i0    = chunk * ROWS_PER_CHUNK;
  float m[4] = {-INFINITY, -INFINITY, -INFINITY, -INFINITY};
  float s[4] = {0.0f, 0.0f, 0.0f, 0.0f};
  const float* col = Mr + (size_t)i0 * MM + j;
  for (int i = 0; i < ROWS_PER_CHUNK; ++i) {
    v4f x = *(const v4f*)(col + (size_t)i * MM);
    const float ui = u[i0 + i];
#pragma unroll
    for (int t = 0; t < 4; ++t) lse_push(m[t], s[t], x[t] + ui);
  }
  v4f om = {m[0], m[1], m[2], m[3]};
  v4f os = {s[0], s[1], s[2], s[3]};
  *(v4f*)(pm + chunk * MM + j) = om;
  *(v4f*)(ps + chunk * MM + j) = os;
}

// K3b: combine chunk partials -> v[j] (4 columns per thread)
__global__ __launch_bounds__(256)
void ot_col_combine(const float* __restrict__ pm, const float* __restrict__ ps,
                    float* __restrict__ v) {
  const int j = (blockIdx.x * 256 + threadIdx.x) * 4;
  float m[4] = {-INFINITY, -INFINITY, -INFINITY, -INFINITY};
  float s[4] = {0.0f, 0.0f, 0.0f, 0.0f};
  for (int c = 0; c < CHUNKS; ++c) {
    v4f mc = *(const v4f*)(pm + c * MM + j);
    v4f sc = *(const v4f*)(ps + c * MM + j);
#pragma unroll
    for (int t = 0; t < 4; ++t) lse_merge(m[t], s[t], mc[t], sc[t]);
  }
  v4f out;
#pragma unroll
  for (int t = 0; t < 4; ++t) out[t] = -LOGNM - (m[t] + __logf(s[t]));
  *(v4f*)(v + j) = out;
}

// ---------------------------------------------------------------------------
// K4: row-logsumexp -> u[i]; one 256-thread block per row, b128 loads,
// per-thread 4-lane lse then LDS (m,s) tree reduce.
// ---------------------------------------------------------------------------
__global__ __launch_bounds__(256)
void ot_row_update(const float* __restrict__ Mr, const float* __restrict__ v,
                   float* __restrict__ u) {
  const int i   = blockIdx.x;
  const int tid = threadIdx.x;
  const float* row = Mr + (size_t)i * MM;
  float m[4] = {-INFINITY, -INFINITY, -INFINITY, -INFINITY};
  float s[4] = {0.0f, 0.0f, 0.0f, 0.0f};
  for (int k = 0; k < MM / 1024; ++k) {           // 8 iterations
    const int j = k * 1024 + tid * 4;
    v4f x  = *(const v4f*)(row + j);
    v4f vj = *(const v4f*)(v + j);
#pragma unroll
    for (int t = 0; t < 4; ++t) lse_push(m[t], s[t], x[t] + vj[t]);
  }
  lse_merge(m[0], s[0], m[1], s[1]);
  lse_merge(m[2], s[2], m[3], s[3]);
  lse_merge(m[0], s[0], m[2], s[2]);

  __shared__ float sm[256], ss[256];
  sm[tid] = m[0]; ss[tid] = s[0];
  __syncthreads();
  for (int off = 128; off; off >>= 1) {
    if (tid < off) {
      float m1 = sm[tid], s1 = ss[tid];
      lse_merge(m1, s1, sm[tid + off], ss[tid + off]);
      sm[tid] = m1; ss[tid] = s1;
    }
    __syncthreads();
  }
  if (tid == 0) u[i] = -LOGNM - (sm[0] + __logf(ss[0]));
}

// ---------------------------------------------------------------------------
// K5: fused finalize + mapped GEMM.
// Block = 16 output rows; 8 waves cover d = 0..127 (16 each).
// For each 16x32 K-tile: read Mr, compute pi = exp(Mr+u+v), write pi to
// d_out AND stage bf16 pi in LDS; then every wave does one WMMA against
// its XQt B-fragment. Mr/pi crosses HBM exactly once here.
// ---------------------------------------------------------------------------
__global__ __launch_bounds__(256)
void ot_mapped_pi_gemm(float* __restrict__ MrPi,           // in: Mr, out: pi
                       const float* __restrict__ u, const float* __restrict__ v,
                       const unsigned short* __restrict__ XQt,
                       float* __restrict__ mapped) {
  const int i0   = blockIdx.x * 16;
  const int tid  = threadIdx.x;
  const int lane = tid & 31;
  const int d0   = (tid >> 5) * 16;       // wave's d-tile
  const int half = lane >> 4;
  const int koff = half * 8;
  const int krng = half * 16;
  const int arow = lane & 15;
  const int dcol = d0 + (lane & 15);

  // cooperative tile-load coordinates (2 consecutive elements per thread)
  const int e0 = tid * 2;
  const int tr = e0 >> 5;                  // 0..15
  const int tc = e0 & 31;                  // even
  const float ur = u[i0 + tr];             // fixed per thread

  __shared__ __align__(16) unsigned short Abuf[16][40];   // 16x32 bf16, padded

  v8f acc = {};
  for (int kb = 0; kb < MM; kb += 32) {
    float* mp = MrPi + (size_t)(i0 + tr) * MM + kb + tc;
    v2f mr = *(const v2f*)mp;
    v2f vj = *(const v2f*)(v + kb + tc);
    float px = __expf(mr.x + ur + vj.x);
    float py = __expf(mr.y + ur + vj.y);
    v2f p = {px, py};
    *(v2f*)mp = p;                         // pi -> d_out (in place over Mr)
    Abuf[tr][tc]     = f2bfu(px);
    Abuf[tr][tc + 1] = f2bfu(py);
    __syncthreads();

    ABu a, b;
    a.q[0] = *(const v4u*)(&Abuf[arow][krng]);
    a.q[1] = *(const v4u*)(&Abuf[arow][krng + 8]);
    const unsigned short* bp = XQt + (size_t)dcol * MM + kb + krng;
    b.q[0] = *(const v4u*)(bp);
    b.q[1] = *(const v4u*)(bp + 8);
    acc = __builtin_amdgcn_wmma_f32_16x16x32_bf16(false, a.bf, false, b.bf,
                                                  (short)0, acc, false, false);
    __syncthreads();
  }
  const int n     = lane & 15;
  const int mbase = half * 8;
#pragma unroll
  for (int r = 0; r < 8; ++r)
    mapped[(size_t)(i0 + mbase + r) * DD + d0 + n] = acc[r] * (float)NN;
}

// ---------------------------------------------------------------------------
extern "C" void kernel_launch(void* const* d_in, const int* in_sizes, int n_in,
                              void* d_out, int out_size, void* d_ws, size_t ws_size,
                              hipStream_t stream) {
  const float* XP = (const float*)d_in[0];
  const float* XQ = (const float*)d_in[1];

  float* mapped = (float*)d_out;                     // [NN*DD]
  float* piBuf  = mapped + (size_t)NN * DD;          // [NN*MM], holds Mr then pi

  // workspace carve-up (all 16B aligned)
  float* xx = (float*)d_ws;                          // NN
  float* yy = xx + NN;                               // MM
  float* u  = yy + MM;                               // NN
  float* vv = u + NN;                                // MM
  float* pm = vv + MM;                               // CHUNKS*MM
  float* ps = pm + (size_t)CHUNKS * MM;              // CHUNKS*MM
  unsigned short* XPb = (unsigned short*)(ps + (size_t)CHUNKS * MM); // NN*DD
  unsigned short* XQb = XPb + (size_t)NN * DD;       // MM*DD
  unsigned short* XQt = XQb + (size_t)MM * DD;       // DD*MM

  ot_prep<<<NN + MM, 128, 0, stream>>>(XP, XQ, xx, yy, u, vv, XPb, XQb, XQt);

  ot_cost_gemm<<<(NN / 16) * (MM / 64) / 8, 256, 0, stream>>>(XPb, XQb, xx, yy, piBuf);

  for (int it = 0; it < N_ITER; ++it) {
    ot_col_partial<<<dim3(MM / 1024, CHUNKS), 256, 0, stream>>>(piBuf, u, pm, ps);
    ot_col_combine<<<MM / 1024, 256, 0, stream>>>(pm, ps, vv);
    ot_row_update<<<NN, 256, 0, stream>>>(piBuf, vv, u);
  }

  ot_mapped_pi_gemm<<<NN / 16, 256, 0, stream>>>(piBuf, u, vv, XQt, mapped);
}
